// Attention_20564303413273
// MI455X (gfx1250) — matrix operand order repaired and checked by reading, not compiled, not run on hardware
//
#include <hip/hip_runtime.h>

#define DIN  1024
#define DOUT 1024
#define TSEQ 2048
#define NB   4
#define SROW 2065   // 16 rows * 2065 f32 in LDS; stride%64=17 -> conflict-free frags

typedef __attribute__((ext_vector_type(16))) _Float16 v16h;
typedef __attribute__((ext_vector_type(8)))  _Float16 v8h;
typedef __attribute__((ext_vector_type(8)))  float    v8f;

union V16H { v16h v; v8h h[2]; };

__device__ __forceinline__ v8f wmma_f16(v16h a, v16h b, v8f c) {
  // D = A(16x32 f16) * B(32x16 f16) + C(16x16 f32)
  return __builtin_amdgcn_wmma_f32_16x16x32_f16(false, a, false, b, (short)0, c,
                                                false, false);
}

// ---------------------------------------------------------------------------
// Kernel 0: streaming fp32 -> f16 conversion (bandwidth-bound, ~3us total)
// ---------------------------------------------------------------------------
__global__ __launch_bounds__(256) void cvt_f16_kernel(
    const float* __restrict__ src, _Float16* __restrict__ dst, int n)
{
  const int i = (blockIdx.x * 256 + threadIdx.x) * 8;
  if (i >= n) return;
  __builtin_prefetch(src + i + 8 * 256 * 8, 0, 0);   // global_prefetch_b8
  v8f s = *(const v8f*)(src + i);
  v8h d;
#pragma unroll
  for (int j = 0; j < 8; ++j) d[j] = (_Float16)s[j];
  *(v8h*)(dst + i) = d;
}

// ---------------------------------------------------------------------------
// Kernel 1: Q = x Wq^T, K = x Wk^T, V = x Wv^T  (all-f16 WMMA, f32 accum)
// Double-buffered fragments with NAMED registers (no runtime-indexed arrays,
// which AMDGPUPromoteAlloca would demote to LDS).
// ---------------------------------------------------------------------------
__global__ __launch_bounds__(128) void qkv_proj_kernel(
    const _Float16* __restrict__ Xh,
    const _Float16* __restrict__ Wqh, const _Float16* __restrict__ Wkh,
    const _Float16* __restrict__ Wvh,
    _Float16* __restrict__ Qh, _Float16* __restrict__ Kh,
    _Float16* __restrict__ Vt)
{
  const int lane = threadIdx.x & 31;
  const int wave = threadIdx.x >> 5;
  const int hi = lane >> 4;
  const int ln = lane & 15;
  const int row0 = blockIdx.x * 16;                 // global token row
  const int col0 = blockIdx.y * 256 + wave * 64;    // output feature col

  v8f accQ[4] = {}, accK[4] = {}, accV[4] = {};

  V16H Acur, Anxt;
  v16h Bq[2], Bk[2], Bv[2];   // constant-indexed under #pragma unroll -> regs

  auto loadA = [&](int k0, V16H& dst) {
    const _Float16* ap = Xh + (size_t)(row0 + ln) * DIN + k0 + hi * 16;
    dst.h[0] = *(const v8h*)ap;
    dst.h[1] = *(const v8h*)(ap + 8);
  };
  auto loadB = [&](int k0, int t, int s) {
    const size_t wrow = (size_t)(col0 + t * 16 + ln) * DIN + k0 + hi * 16;
    Bq[s] = *(const v16h*)(Wqh + wrow);
    Bk[s] = *(const v16h*)(Wkh + wrow);
    Bv[s] = *(const v16h*)(Wvh + wrow);
  };

  loadA(0, Acur);
  loadB(0, 0, 0);
  for (int k0 = 0; k0 < DIN; k0 += 32) {
#pragma unroll
    for (int t = 0; t < 4; ++t) {
      const int cs = t & 1, ns = cs ^ 1;
      if (t < 3) {
        loadB(k0, t + 1, ns);                 // prefetch next tile's fragments
      } else if (k0 + 32 < DIN) {
        loadA(k0 + 32, Anxt);                 // prefetch next k-chunk's A
        loadB(k0 + 32, 0, ns);                // t=3 loads slot 0 -> next t=0
      }
      accQ[t] = wmma_f16(Acur.v, Bq[cs], accQ[t]);
      accK[t] = wmma_f16(Acur.v, Bk[cs], accK[t]);
      accV[t] = wmma_f16(Acur.v, Bv[cs], accV[t]);
    }
    Acur.v = Anxt.v;                          // rotate (reg moves, no memory)
  }

#pragma unroll
  for (int t = 0; t < 4; ++t) {
    const int n = col0 + t * 16 + ln;
#pragma unroll
    for (int i = 0; i < 8; ++i) {
      const int r = row0 + i + hi * 8;            // global token row
      Qh[(size_t)r * DOUT + n] = (_Float16)accQ[t][i];
      Kh[(size_t)r * DOUT + n] = (_Float16)accK[t][i];
      const int bb = r >> 11;                     // r / TSEQ
      const int tl = r & (TSEQ - 1);              // r % TSEQ
      Vt[((size_t)bb * DOUT + n) * TSEQ + tl] = (_Float16)accV[t][i];
    }
  }
}

// ---------------------------------------------------------------------------
// Kernel 2 (fused): S = QK^T (tril mask * 1/sqrt(T)), row softmax, O = P V
// One workgroup (256 thr = 8 waves) per 16 query rows per batch.
// LDS: 16 x SROW f32 + reduction scratch (~130 KB, < 320 KB/WGP).
// ---------------------------------------------------------------------------
__global__ __launch_bounds__(256) void attn_kernel(
    const _Float16* __restrict__ Qh, const _Float16* __restrict__ Kh,
    const _Float16* __restrict__ Vt, float* __restrict__ out)
{
  extern __shared__ float smem[];
  float* red  = smem + 16 * SROW;   // 256 partials
  float* rowM = red + 256;          // 16 row maxes
  float* rowL = rowM + 16;          // 16 row sums

  const int tid  = threadIdx.x;
  const int lane = tid & 31;
  const int wave = tid >> 5;        // 0..7
  const int hi = lane >> 4, ln = lane & 15;
  const int b  = blockIdx.y;
  const int q0 = blockIdx.x * 16;   // query tile base within batch

  const _Float16* Qb = Qh + (size_t)b * TSEQ * DOUT;
  const _Float16* Kb = Kh + (size_t)b * TSEQ * DOUT;
  const _Float16* Vb = Vt + (size_t)b * DOUT * TSEQ;

  // ---- Phase A: S tile. This wave: key cols [wave*256, wave*256+256)
  {
    v8f accS[16] = {};
    V16H Aqc, Aqn;
    v16h Bk2[2];
    auto loadAq = [&](int k0, V16H& d) {
      const _Float16* qp = Qb + (size_t)(q0 + ln) * DOUT + k0 + hi * 16;
      d.h[0] = *(const v8h*)qp;
      d.h[1] = *(const v8h*)(qp + 8);
    };
    auto loadBk = [&](int k0, int t, int s) {
      const int n = wave * 256 + t * 16 + ln;     // key index (B column)
      Bk2[s] = *(const v16h*)(Kb + (size_t)n * DOUT + k0 + hi * 16);
    };

    loadAq(0, Aqc);
    loadBk(0, 0, 0);
    for (int k0 = 0; k0 < DIN; k0 += 32) {
#pragma unroll
      for (int t = 0; t < 16; ++t) {
        const int cs = t & 1, ns = cs ^ 1;
        if (t < 15) {
          loadBk(k0, t + 1, ns);
        } else if (k0 + 32 < DIN) {
          loadAq(k0 + 32, Aqn);
          loadBk(k0 + 32, 0, ns);               // t=15 loads slot 0 -> next t=0
        }
        accS[t] = wmma_f16(Aqc.v, Bk2[cs], accS[t]);
      }
      Aqc.v = Aqn.v;                            // rotate (reg moves)
    }
    const float rs = 0.02209708691207961f;  // 1/sqrt(2048)
#pragma unroll
    for (int t = 0; t < 16; ++t) {
      const int j = wave * 256 + t * 16 + ln;     // key col
#pragma unroll
      for (int i = 0; i < 8; ++i) {
        const int r = i + hi * 8;                 // local query row
        const int g = q0 + r;                     // global query row
        // multiplicative tril mask: above-diagonal logits become 0 (not -inf)
        smem[r * SROW + j] = (j <= g) ? accS[t][i] * rs : 0.0f;
      }
    }
  }
  __syncthreads();

  // ---- Softmax over each of the 16 rows (2048 cols); 16 threads per row,
  //      stride-16 interleaved columns to spread LDS banks.
  {
    const int r = tid >> 4, sub = tid & 15;
    float m = -3.0e38f;
    for (int i = 0; i < 128; ++i)
      m = fmaxf(m, smem[r * SROW + sub + i * 16]);
    red[tid] = m;
    __syncthreads();
    if (sub == 0) {
      float mm = red[r * 16];
      for (int s = 1; s < 16; ++s) mm = fmaxf(mm, red[r * 16 + s]);
      rowM[r] = mm;
    }
    __syncthreads();
    const float mr = rowM[r];
    float l = 0.0f;
    for (int i = 0; i < 128; ++i) {
      const int c = sub + i * 16;
      const float p = __expf(smem[r * SROW + c] - mr);
      smem[r * SROW + c] = p;
      l += p;
    }
    red[tid] = l;
    __syncthreads();
    if (sub == 0) {
      float ll = 0.0f;
      for (int s = 0; s < 16; ++s) ll += red[r * 16 + s];
      rowL[r] = ll;
    }
    __syncthreads();
    const float inv = 1.0f / rowL[r];
    for (int i = 0; i < 128; ++i)
      smem[r * SROW + sub + i * 16] *= inv;
  }
  __syncthreads();

  // ---- Phase B: O = P V. This wave: dout cols [wave*128, wave*128+128)
  {
    v8f accO[8] = {};
    v16h Bv2[2];
    auto loadBv = [&](int k0, int t, int s) {
      const int n = wave * 128 + t * 16 + ln;     // dout col
      Bv2[s] = *(const v16h*)(Vb + (size_t)n * TSEQ + k0 + hi * 16);
    };

    loadBv(0, 0, 0);
    for (int k0 = 0; k0 < TSEQ; k0 += 32) {
      V16H A;
      const float* pp = smem + ln * SROW + k0 + hi * 16;
#pragma unroll
      for (int j = 0; j < 8; ++j) {
        A.v[j]     = (_Float16)pp[j];
        A.v[8 + j] = (_Float16)pp[8 + j];
      }
#pragma unroll
      for (int t = 0; t < 8; ++t) {
        const int cs = t & 1, ns = cs ^ 1;
        if (t < 7) {
          loadBv(k0, t + 1, ns);
        } else if (k0 + 32 < TSEQ) {
          loadBv(k0 + 32, 0, ns);                 // t=7 loads slot 0 -> next t=0
        }
        accO[t] = wmma_f16(A.v, Bv2[cs], accO[t]);
      }
    }
#pragma unroll
    for (int t = 0; t < 8; ++t) {
      const int n = wave * 128 + t * 16 + ln;
#pragma unroll
      for (int i = 0; i < 8; ++i) {
        out[((size_t)b * TSEQ + q0 + i + hi * 8) * DOUT + n] = accO[t][i];
      }
    }
  }
}

// ---------------------------------------------------------------------------
extern "C" void kernel_launch(void* const* d_in, const int* in_sizes, int n_in,
                              void* d_out, int out_size, void* d_ws, size_t ws_size,
                              hipStream_t stream) {
  const float* x  = (const float*)d_in[0];
  const float* Wq = (const float*)d_in[1];
  const float* Wk = (const float*)d_in[2];
  const float* Wv = (const float*)d_in[3];
  float* out = (float*)d_out;

  const size_t NX = (size_t)NB * TSEQ * DIN;   // 8M elems
  const size_t NW = (size_t)DOUT * DIN;        // 1M elems

  // Workspace layout (f16):
  //   Xh 16MB | Wqh 2MB | Wkh 2MB | Wvh 2MB | Qh 16MB | Kh 16MB | Vt 16MB
  _Float16* Xh  = (_Float16*)d_ws;
  _Float16* Wqh = Xh  + NX;
  _Float16* Wkh = Wqh + NW;
  _Float16* Wvh = Wkh + NW;
  _Float16* Qh  = Wvh + NW;
  _Float16* Kh  = Qh + (size_t)NB * TSEQ * DOUT;
  _Float16* Vt  = Kh + (size_t)NB * TSEQ * DOUT;

  cvt_f16_kernel<<<dim3(NX / (256 * 8)), 256, 0, stream>>>(x,  Xh,  (int)NX);
  cvt_f16_kernel<<<dim3(NW / (256 * 8)), 256, 0, stream>>>(Wq, Wqh, (int)NW);
  cvt_f16_kernel<<<dim3(NW / (256 * 8)), 256, 0, stream>>>(Wk, Wkh, (int)NW);
  cvt_f16_kernel<<<dim3(NW / (256 * 8)), 256, 0, stream>>>(Wv, Wvh, (int)NW);

  qkv_proj_kernel<<<dim3(NB * TSEQ / 16, DOUT / 256), 128, 0, stream>>>(
      Xh, Wqh, Wkh, Wvh, Qh, Kh, Vt);

  const size_t smem_bytes = (size_t)(16 * SROW + 256 + 32) * sizeof(float);
  attn_kernel<<<dim3(TSEQ / 16, NB), 256, smem_bytes, stream>>>(Qh, Kh, Vt, out);
}
